// Coarse2FineGraphEncoder_12962211299773
// MI455X (gfx1250) — matrix operand order repaired and checked by reading, not compiled, not run on hardware
//
#include <hip/hip_runtime.h>

typedef __bf16 bf16;
typedef bf16  bf16x2  __attribute__((ext_vector_type(2)));
typedef bf16  bf16x8  __attribute__((ext_vector_type(8)));
typedef bf16  bf16x16 __attribute__((ext_vector_type(16)));
typedef float f32x8   __attribute__((ext_vector_type(8)));

#define DEV __device__ __forceinline__

DEV f32x8 wmma_bf16(bf16x16 a, bf16x16 b, f32x8 c) {
  return __builtin_amdgcn_wmma_f32_16x16x32_bf16(false, a, false, b, (short)0, c, false, false);
}

// A-fragment: 16x32 bf16 tile, row-major LDS, stride 40 (80B rows, 16B aligned).
// Lane (row, hf) holds K = {hf*8..+7} and {16+hf*8..+7}: 2x ds_load_b128.
DEV bf16x16 frag_a(const bf16* base, int ld, int lane) {
  const int hf = lane >> 4, row = lane & 15;
  const bf16* p = base + row * ld + hf * 8;
  bf16x8 lo = *(const bf16x8*)p;
  bf16x8 hi = *(const bf16x8*)(p + 16);
  return __builtin_shufflevector(lo, hi, 0, 1, 2, 3, 4, 5, 6, 7,
                                 8, 9, 10, 11, 12, 13, 14, 15);
}

// B-fragment from a TRANSPOSED tile Bt[n][k] (K innermost). Lane (col, kb)
// holds K = {kb..kb+7} and {16+kb..+7} of column col: 2x ds_load_b128.
DEV bf16x16 frag_bt(const bf16* base, int ld, int lane) {
  const int col = lane & 15, kb = (lane >> 4) * 8;
  const bf16* p = base + col * ld + kb;
  bf16x8 lo = *(const bf16x8*)p;
  bf16x8 hi = *(const bf16x8*)(p + 16);
  return __builtin_shufflevector(lo, hi, 0, 1, 2, 3, 4, 5, 6, 7,
                                 8, 9, 10, 11, 12, 13, 14, 15);
}

DEV void store_pair(bf16* dst, float a, float b) {
  bf16x2 p; p[0] = (bf16)a; p[1] = (bf16)b;   // v_cvt_pk_bf16_f32 -> ds_store_b32
  *(bf16x2*)dst = p;
}

// ---------------------------------------------------------------------------
// Generic GEMM: C[M,N] = act(A[M,K] @ W[K,N] + bias).  N,K compile-time.
// Block 256 thr = 8 waves; wave computes 16x32 (2 WMMA C-frags).
// Double-buffered LDS with register staging: global fetch of tile kt+1
// overlaps WMMA on tile kt.
// ---------------------------------------------------------------------------
template <int N, int K, bool BIAS, bool RELU>
__global__ __launch_bounds__(256)
void k_gemm(const float* __restrict__ A, const float* __restrict__ W,
            const float* __restrict__ bias, float* __restrict__ C)
{
  constexpr int KT = K / 32;
  __shared__ alignas(16) bf16 As[2][64][40];   // [m][k]
  __shared__ alignas(16) bf16 Bt[2][64][40];   // [n][k] (transposed W tile)
  const int tid = threadIdx.x, lane = tid & 31, wave = tid >> 5;
  const int wm = wave & 3, wn = wave >> 2;
  const int bm = blockIdx.y * 64, bn = blockIdx.x * 64;

  // static loader index decomposition
  const int am = tid >> 4, ak2 = (tid & 15) * 2;   // A: rows am+16*it, k-pair ak2
  const int bnn = tid & 63, bk = tid >> 6;         // W: col bnn, k-pairs bk+4*it

  float2 ra[4];
  float  rw0[4], rw1[4];

  auto fetch = [&](int k0) {
#pragma unroll
    for (int it = 0; it < 4; ++it)
      ra[it] = *(const float2*)(A + (size_t)(bm + am + it * 16) * K + k0 + ak2);
#pragma unroll
    for (int it = 0; it < 4; ++it) {
      int k = k0 + 2 * (bk + it * 4);
      rw0[it] = W[(size_t)k * N + bn + bnn];
      rw1[it] = W[(size_t)(k + 1) * N + bn + bnn];
    }
  };
  auto stage = [&](int buf) {
#pragma unroll
    for (int it = 0; it < 4; ++it)
      store_pair(&As[buf][am + it * 16][ak2], ra[it].x, ra[it].y);
#pragma unroll
    for (int it = 0; it < 4; ++it)
      store_pair(&Bt[buf][bnn][2 * (bk + it * 4)], rw0[it], rw1[it]);
  };

  f32x8 acc0 = {}, acc1 = {};
  fetch(0);
  stage(0);
  __syncthreads();
#pragma unroll 2
  for (int kt = 0; kt < KT; ++kt) {
    if (kt + 1 < KT) fetch((kt + 1) * 32);
    const int cur = kt & 1;
    bf16x16 af = frag_a(&As[cur][wm * 16][0], 40, lane);
    bf16x16 b0 = frag_bt(&Bt[cur][wn * 32][0], 40, lane);
    bf16x16 b1 = frag_bt(&Bt[cur][wn * 32 + 16][0], 40, lane);
    acc0 = wmma_bf16(af, b0, acc0);
    acc1 = wmma_bf16(af, b1, acc1);
    if (kt + 1 < KT) stage((kt + 1) & 1);
    __syncthreads();
  }
  const int hf = lane >> 4, cn = lane & 15;
#pragma unroll
  for (int i = 0; i < 8; ++i) {
    int r  = bm + wm * 16 + i + hf * 8;
    int c0 = bn + wn * 32 + cn;
    int c1 = c0 + 16;
    float v0 = acc0[i], v1 = acc1[i];
    if constexpr (BIAS) { v0 += bias[c0]; v1 += bias[c1]; }
    if constexpr (RELU) { v0 = fmaxf(v0, 0.f); v1 = fmaxf(v1, 0.f); }
    C[(size_t)r * N + c0] = v0;
    C[(size_t)r * N + c1] = v1;
  }
}

// ---------------------------------------------------------------------------
// Embedding gather + input projection: x = relu(embed[tok] @ W_in + b_in)
// ---------------------------------------------------------------------------
__global__ __launch_bounds__(256)
void k_embed(const int* __restrict__ tok, const float* __restrict__ emb,
             const float* __restrict__ Wi, const float* __restrict__ bi,
             float* __restrict__ x)
{
  __shared__ float e[300];
  const int row = blockIdx.x;
  const int t = tok[row];
  for (int i = threadIdx.x; i < 300; i += 256) e[i] = emb[(size_t)t * 300 + i];
  __syncthreads();
  const int c = threadIdx.x;
  float s = bi[c];
#pragma unroll 4
  for (int k = 0; k < 300; ++k) s = fmaf(e[k], Wi[(size_t)k * 256 + c], s);
  x[(size_t)row * 256 + c] = fmaxf(s, 0.f);
}

// ---------------------------------------------------------------------------
// qr[b,h,n,r] = q[b,n,h*32:] . rel_k_emb[r,:]   (RAT-SQL rel-score factorization)
// ---------------------------------------------------------------------------
__global__ __launch_bounds__(256)
void k_qr(const float* __restrict__ q, const float* __restrict__ relk,
          float* __restrict__ qr)
{
  size_t idx = (size_t)blockIdx.x * 256 + threadIdx.x;
  const int r = (int)(idx % 100);
  size_t t = idx / 100;
  const int n = (int)(t & 511); t >>= 9;
  const int h = (int)(t & 7);
  const int b = (int)(t >> 3);
  const float4* qp = (const float4*)(q + ((size_t)(b * 512) + n) * 256 + h * 32);
  const float4* rp = (const float4*)(relk + r * 32);
  float s = 0.f;
#pragma unroll
  for (int d = 0; d < 8; ++d) {
    float4 a = qp[d], bb = rp[d];
    s += a.x * bb.x + a.y * bb.y + a.z * bb.z + a.w * bb.w;
  }
  qr[idx] = s;
}

// ---------------------------------------------------------------------------
// scores[b,h,n,m] = (q.k + qr[b,h,n,rel[b,n,m]]) * scale, masked to -1e9.
// 64x64 tile per block per (b,h); K = dh = 32 (single WMMA step).
// ---------------------------------------------------------------------------
__global__ __launch_bounds__(256)
void k_scores(const float* __restrict__ q, const float* __restrict__ km,
              const int* __restrict__ rel, const unsigned char* __restrict__ msk,
              const float* __restrict__ qr, float* __restrict__ sc)
{
  __shared__ alignas(16) bf16 Qs[64][40];   // [n][d]
  __shared__ alignas(16) bf16 Kt[64][40];   // [m][d]  (natural == Bt layout)
  const int tid = threadIdx.x, lane = tid & 31, wave = tid >> 5;
  const int wm = wave & 3, wn = wave >> 2;
  const int bh = blockIdx.z, b = bh >> 3, h = bh & 7;
  const int n0 = blockIdx.y * 64, m0 = blockIdx.x * 64;
  const int lr = tid >> 4, d2 = (tid & 15) * 2;

  float2 rq[4], rk[4];
#pragma unroll
  for (int it = 0; it < 4; ++it)
    rq[it] = *(const float2*)(q + ((size_t)(b * 512) + n0 + lr + it * 16) * 256 + h * 32 + d2);
#pragma unroll
  for (int it = 0; it < 4; ++it)
    rk[it] = *(const float2*)(km + ((size_t)(b * 512) + m0 + lr + it * 16) * 256 + h * 32 + d2);
#pragma unroll
  for (int it = 0; it < 4; ++it) store_pair(&Qs[lr + it * 16][d2], rq[it].x, rq[it].y);
#pragma unroll
  for (int it = 0; it < 4; ++it) store_pair(&Kt[lr + it * 16][d2], rk[it].x, rk[it].y);
  __syncthreads();

  bf16x16 af = frag_a(&Qs[wm * 16][0], 40, lane);
  bf16x16 b0 = frag_bt(&Kt[wn * 32][0], 40, lane);
  bf16x16 b1 = frag_bt(&Kt[wn * 32 + 16][0], 40, lane);
  f32x8 acc0 = {}, acc1 = {};
  acc0 = wmma_bf16(af, b0, acc0);
  acc1 = wmma_bf16(af, b1, acc1);

  const float scale = 0.17677669529663687f;  // 1/sqrt(32)
  const int hf = lane >> 4, cn = lane & 15;
  const float* qrow = qr + (size_t)bh * 512 * 100;
#pragma unroll
  for (int i = 0; i < 8; ++i) {
    int n  = n0 + wm * 16 + i + hf * 8;
    int mA = m0 + wn * 32 + cn;
    int mB = mA + 16;
    size_t relbase = ((size_t)(b * 512) + n) * 512;
    int rA = rel[relbase + mA];
    int rB = rel[relbase + mB];
    float sA = (acc0[i] + qrow[(size_t)n * 100 + rA]) * scale;
    float sB = (acc1[i] + qrow[(size_t)n * 100 + rB]) * scale;
    if (!msk[relbase + mA]) sA = -1e9f;
    if (!msk[relbase + mB]) sB = -1e9f;
    size_t srow = ((size_t)bh * 512 + n) * 512;
    sc[srow + mA] = sA;
    sc[srow + mB] = sB;
  }
}

// ---------------------------------------------------------------------------
// Row softmax (in place) + relation bucket accumulation (LDS ds_add_f32):
// ar[b,h,n,r] = sum_{m: rel[b,n,m]==r} p[b,h,n,m].  One wave per row.
// ---------------------------------------------------------------------------
__global__ __launch_bounds__(256)
void k_softmax_ar(float* __restrict__ sc, const int* __restrict__ rel,
                  float* __restrict__ ar)
{
  __shared__ float buckets[8][104];
  const int lane = threadIdx.x & 31, wave = threadIdx.x >> 5;
  const size_t row = (size_t)blockIdx.x * 8 + wave;  // (b*8+h)*512 + n
  const int n = (int)(row & 511);
  const int b = (int)(row >> 12);
  float* srow = sc + row * 512;
  const int* rrow = rel + ((size_t)(b * 512) + n) * 512;
  for (int i = lane; i < 104; i += 32) buckets[wave][i] = 0.f;
  float4 v4[4];
  float mx = -3.0e38f;
#pragma unroll
  for (int j = 0; j < 4; ++j) {
    v4[j] = *(const float4*)(srow + lane * 16 + j * 4);
    mx = fmaxf(mx, fmaxf(fmaxf(v4[j].x, v4[j].y), fmaxf(v4[j].z, v4[j].w)));
  }
#pragma unroll
  for (int off = 16; off > 0; off >>= 1) mx = fmaxf(mx, __shfl_xor(mx, off, 32));
  float sum = 0.f;
#pragma unroll
  for (int j = 0; j < 4; ++j) {
    v4[j].x = __expf(v4[j].x - mx); v4[j].y = __expf(v4[j].y - mx);
    v4[j].z = __expf(v4[j].z - mx); v4[j].w = __expf(v4[j].w - mx);
    sum += v4[j].x + v4[j].y + v4[j].z + v4[j].w;
  }
#pragma unroll
  for (int off = 16; off > 0; off >>= 1) sum += __shfl_xor(sum, off, 32);
  const float inv = 1.f / sum;
#pragma unroll
  for (int j = 0; j < 4; ++j) {
    int4 r4 = *(const int4*)(rrow + lane * 16 + j * 4);
    v4[j].x *= inv; v4[j].y *= inv; v4[j].z *= inv; v4[j].w *= inv;
    *(float4*)(srow + lane * 16 + j * 4) = v4[j];
    atomicAdd(&buckets[wave][r4.x], v4[j].x);
    atomicAdd(&buckets[wave][r4.y], v4[j].y);
    atomicAdd(&buckets[wave][r4.z], v4[j].z);
    atomicAdd(&buckets[wave][r4.w], v4[j].w);
  }
  __syncthreads();
  for (int r = lane; r < 100; r += 32) ar[row * 100 + r] = buckets[wave][r];
}

// ---------------------------------------------------------------------------
// z[b,n,h*32+d] = P @ V  +  ar @ rel_v_emb   (per (b,h), 64-row tiles).
// Double-buffered main loop over m (16 k-steps); rel term K=100 padded to 128.
// ---------------------------------------------------------------------------
__global__ __launch_bounds__(256)
void k_pv(const float* __restrict__ p, const float* __restrict__ v,
          const float* __restrict__ ar, const float* __restrict__ relv,
          float* __restrict__ z)
{
  __shared__ alignas(16) bf16 Ps[2][64][40];   // [n][k]
  __shared__ alignas(16) bf16 Vt[2][32][40];   // [d][k]
  const int tid = threadIdx.x, lane = tid & 31, wave = tid >> 5;
  const int wm = wave & 3, wn = wave >> 2;
  const int bh = blockIdx.y, b = bh >> 3, h = bh & 7;
  const int n0 = blockIdx.x * 64;
  const int pn = tid >> 4, pm2 = (tid & 15) * 2;   // Ps: rows pn+16*it, pair pm2
  const int vd = tid & 31, vm = tid >> 5;          // Vt: col vd, m-pairs vm+8*it
  const float* prow = p + ((size_t)bh * 512 + n0) * 512;
  const size_t vbase = (size_t)(b * 512) * 256 + h * 32 + vd;

  float2 rp_[4];
  float  rv0[2], rv1[2];
  auto fetch = [&](int m0) {
#pragma unroll
    for (int it = 0; it < 4; ++it)
      rp_[it] = *(const float2*)(prow + (size_t)(pn + it * 16) * 512 + m0 + pm2);
#pragma unroll
    for (int it = 0; it < 2; ++it) {
      size_t vb = vbase + (size_t)(m0 + 2 * (vm + it * 8)) * 256;
      rv0[it] = v[vb];
      rv1[it] = v[vb + 256];
    }
  };
  auto stage = [&](int buf) {
#pragma unroll
    for (int it = 0; it < 4; ++it)
      store_pair(&Ps[buf][pn + it * 16][pm2], rp_[it].x, rp_[it].y);
#pragma unroll
    for (int it = 0; it < 2; ++it)
      store_pair(&Vt[buf][vd][2 * (vm + it * 8)], rv0[it], rv1[it]);
  };

  f32x8 acc = {};
  fetch(0);
  stage(0);
  __syncthreads();
#pragma unroll 2
  for (int kt = 0; kt < 16; ++kt) {
    if (kt + 1 < 16) fetch((kt + 1) * 32);
    const int cur = kt & 1;
    bf16x16 af  = frag_a(&Ps[cur][wm * 16][0], 40, lane);
    bf16x16 bfr = frag_bt(&Vt[cur][wn * 16][0], 40, lane);
    acc = wmma_bf16(af, bfr, acc);
    if (kt + 1 < 16) stage((kt + 1) & 1);
    __syncthreads();
  }

  // relation-value term: K over R=100 padded to 128
  const float* arow = ar + ((size_t)bh * 512 + n0) * 100;
  for (int r0 = 0; r0 < 128; r0 += 32) {
    {
      int r = r0 + pm2;
#pragma unroll
      for (int it = 0; it < 4; ++it) {
        if (r < 100) {  // R=100 even: pair never straddles
          float2 f = *(const float2*)(arow + (size_t)(pn + it * 16) * 100 + r);
          rp_[it] = f;
        } else {
          rp_[it].x = 0.f; rp_[it].y = 0.f;
        }
      }
#pragma unroll
      for (int it = 0; it < 2; ++it) {
        int rr = r0 + 2 * (vm + it * 8);
        rv0[it] = (rr < 100) ? relv[rr * 32 + vd] : 0.f;
        rv1[it] = (rr < 100) ? relv[(rr + 1) * 32 + vd] : 0.f;
      }
    }
    stage(0);
    __syncthreads();
    bf16x16 af  = frag_a(&Ps[0][wm * 16][0], 40, lane);
    bf16x16 bfr = frag_bt(&Vt[0][wn * 16][0], 40, lane);
    acc = wmma_bf16(af, bfr, acc);
    __syncthreads();
  }

  const int hf = lane >> 4, cn = lane & 15;
#pragma unroll
  for (int i = 0; i < 8; ++i) {
    int n = n0 + wm * 16 + i + hf * 8;
    int d = wn * 16 + cn;
    z[((size_t)(b * 512) + n) * 256 + h * 32 + d] = acc[i];
  }
}

// ---------------------------------------------------------------------------
// out = LayerNorm(resid + y) * g + b   (D=256, one block per row)
// wave shfl reductions + 8 LDS partials (2 barriers total)
// ---------------------------------------------------------------------------
__global__ __launch_bounds__(256)
void k_add_ln(const float* __restrict__ resid, const float* __restrict__ y,
              const float* __restrict__ g, const float* __restrict__ bta,
              float* __restrict__ out)
{
  __shared__ float red[8], red2[8];
  const int row = blockIdx.x, c = threadIdx.x;
  const int lane = c & 31, wave = c >> 5;
  float vv = resid[(size_t)row * 256 + c] + y[(size_t)row * 256 + c];
  float s = vv;
#pragma unroll
  for (int off = 16; off > 0; off >>= 1) s += __shfl_xor(s, off, 32);
  if (lane == 0) red[wave] = s;
  __syncthreads();
  float tot = 0.f;
#pragma unroll
  for (int i = 0; i < 8; ++i) tot += red[i];
  float mu = tot * (1.f / 256.f);
  float d = vv - mu;
  float s2 = d * d;
#pragma unroll
  for (int off = 16; off > 0; off >>= 1) s2 += __shfl_xor(s2, off, 32);
  if (lane == 0) red2[wave] = s2;
  __syncthreads();
  float tv = 0.f;
#pragma unroll
  for (int i = 0; i < 8; ++i) tv += red2[i];
  float var = tv * (1.f / 256.f);
  out[(size_t)row * 256 + c] = d * rsqrtf(var + 1e-5f) * g[c] + bta[c];
}

// ---------------------------------------------------------------------------
extern "C" void kernel_launch(void* const* d_in, const int* in_sizes, int n_in,
                              void* d_out, int out_size, void* d_ws, size_t ws_size,
                              hipStream_t stream)
{
  (void)in_sizes; (void)n_in; (void)out_size; (void)ws_size;
  const int*   tok  = (const int*)d_in[0];
  const int*   rel  = (const int*)d_in[1];
  const unsigned char* msk = (const unsigned char*)d_in[2];   // jnp bool -> 1 byte
  const float* emb  = (const float*)d_in[3];
  const float* W_in = (const float*)d_in[4];
  const float* b_in = (const float*)d_in[5];
  const float* Wq   = (const float*)d_in[6];
  const float* Wk   = (const float*)d_in[7];
  const float* Wv   = (const float*)d_in[8];
  const float* Wo   = (const float*)d_in[9];
  const float* ln1g = (const float*)d_in[10];
  const float* ln1b = (const float*)d_in[11];
  const float* ln2g = (const float*)d_in[12];
  const float* ln2b = (const float*)d_in[13];
  const float* W1   = (const float*)d_in[14];
  const float* b1   = (const float*)d_in[15];
  const float* W2   = (const float*)d_in[16];
  const float* b2   = (const float*)d_in[17];
  const float* relk = (const float*)d_in[18];
  const float* relv = (const float*)d_in[19];
  const float* Wout = (const float*)d_in[20];
  const float* bout = (const float*)d_in[21];
  float* out = (float*)d_out;

  // workspace layout (floats)
  float* ws = (float*)d_ws;
  const size_t XSZ = (size_t)4096 * 256;
  float* xA = ws;
  float* xB = xA + XSZ;
  float* qb = xB + XSZ;
  float* kb = qb + XSZ;
  float* vb = kb + XSZ;
  float* tb = vb + XSZ;
  float* hb = tb + XSZ;                       // 4096 x 1024
  float* qr = hb + (size_t)4096 * 1024;       // 32768 x 100
  float* ar = qr + (size_t)32768 * 100;       // 32768 x 100
  float* sc = ar + (size_t)32768 * 100;       // 64 x 512 x 512

  const dim3 blk(256);
  k_embed<<<4096, blk, 0, stream>>>(tok, emb, W_in, b_in, xA);

  float* xcur = xA;
  float* xalt = xB;
  for (int l = 0; l < 4; ++l) {
    const size_t wDD = (size_t)l * 256 * 256;
    // QKV projections (WMMA GEMMs)
    k_gemm<256, 256, false, false><<<dim3(4, 64), blk, 0, stream>>>(xcur, Wq + wDD, nullptr, qb);
    k_gemm<256, 256, false, false><<<dim3(4, 64), blk, 0, stream>>>(xcur, Wk + wDD, nullptr, kb);
    k_gemm<256, 256, false, false><<<dim3(4, 64), blk, 0, stream>>>(xcur, Wv + wDD, nullptr, vb);
    // relation-score factorization
    k_qr<<<12800, blk, 0, stream>>>(qb, relk, qr);
    // scores = q.k + qr-gather, masked
    k_scores<<<dim3(8, 8, 64), blk, 0, stream>>>(qb, kb, rel, msk, qr, sc);
    // softmax + relation bucket accumulation
    k_softmax_ar<<<4096, blk, 0, stream>>>(sc, rel, ar);
    // z = P@V + ar@rel_v_emb   (reuse qb as z)
    k_pv<<<dim3(8, 64), blk, 0, stream>>>(sc, vb, ar, relv, qb);
    // output proj + residual LN
    k_gemm<256, 256, false, false><<<dim3(4, 64), blk, 0, stream>>>(qb, Wo + wDD, nullptr, tb);
    k_add_ln<<<4096, blk, 0, stream>>>(xcur, tb, ln1g + l * 256, ln1b + l * 256, xalt);
    { float* t = xcur; xcur = xalt; xalt = t; }
    // FFN
    k_gemm<1024, 256, true, true><<<dim3(16, 64), blk, 0, stream>>>(
        xcur, W1 + (size_t)l * 256 * 1024, b1 + l * 1024, hb);
    k_gemm<256, 1024, true, false><<<dim3(4, 64), blk, 0, stream>>>(
        hb, W2 + (size_t)l * 1024 * 256, b2 + l * 256, tb);
    k_add_ln<<<4096, blk, 0, stream>>>(xcur, tb, ln2g + l * 256, ln2b + l * 256, xalt);
    { float* t = xcur; xcur = xalt; xalt = t; }
  }
  // final projection straight into d_out
  k_gemm<256, 256, true, false><<<dim3(4, 64), blk, 0, stream>>>(xcur, Wout, bout, out);
}
